// GraphConvolutionLayer_88227218194773
// MI455X (gfx1250) — compile-verified, matrix operand order checked by reading, not run on hardware
//
#include <hip/hip_runtime.h>

typedef __attribute__((ext_vector_type(16))) _Float16 v16h;
typedef __attribute__((ext_vector_type(8)))  float    v8f;

// Problem constants (match reference setup_inputs).
constexpr int N     = 50000;
constexpr int R     = 4;
constexpr int E     = 800000;
constexpr int D_IN  = 128;
constexpr int D_OUT = 64;
constexpr int NT    = (N + 15) / 16;      // 3125 row tiles of 16

// ---------------------------------------------------------------------------
// Pack x (f32, row-major [N,128]) into WMMA A-fragments (f16).
// A 16x32 f16 layout (ISA 7.12.2): lane l holds row M = l&15;
//   halves 0..7  -> K = kt*32 + (l>>4)*8 + j
//   halves 8..15 -> K = kt*32 + 16 + (l>>4)*8 + (j-8)
// Fragment storage: [tile][kt][lane][16 halves], contiguous.
// ---------------------------------------------------------------------------
__global__ void pack_x_kernel(const float* __restrict__ x, _Float16* __restrict__ xf) {
    int idx = blockIdx.x * blockDim.x + threadIdx.x;     // NT*4*32*16 = 6,400,000
    if (idx >= NT * 2048) return;
    int j    = idx & 15;
    int lane = (idx >> 4) & 31;
    int kt   = (idx >> 9) & 3;
    int tile = idx >> 11;
    int row  = tile * 16 + (lane & 15);
    int koff = (lane >> 4) * 8;
    int K    = kt * 32 + ((j < 8) ? (koff + j) : (16 + koff + (j - 8)));
    float v  = (row < N) ? x[(size_t)row * D_IN + K] : 0.0f;
    xf[idx]  = (_Float16)v;
}

// Pack W (f32, [R,128,64]) into WMMA B-fragments (f16), mirrored layout:
// lane l holds column Nc = nt*16 + (l&15), same K pattern as A.
// Storage: [r][kt][nt][lane][16 halves]; per-relation stride = 4*4*32*16 = 8192 halves.
__global__ void pack_w_kernel(const float* __restrict__ W, _Float16* __restrict__ wf) {
    int idx = blockIdx.x * blockDim.x + threadIdx.x;     // R*4*4*32*16 = 32768
    if (idx >= R * 8192) return;
    int j    = idx & 15;
    int lane = (idx >> 4) & 31;
    int nt   = (idx >> 9) & 3;
    int kt   = (idx >> 11) & 3;
    int r    = idx >> 13;
    int n    = nt * 16 + (lane & 15);
    int koff = (lane >> 4) * 8;
    int K    = kt * 32 + ((j < 8) ? (koff + j) : (16 + koff + (j - 8)));
    wf[idx]  = (_Float16)W[((size_t)r * D_IN + K) * D_OUT + n];
}

// ---------------------------------------------------------------------------
// GEMM: xw_r[N,64] = x[N,128] @ W[r][128,64] via v_wmma_f32_16x16x32_f16.
// One wave per 16-row tile; 4 accumulators cover all 64 output columns;
// K loop = 4 steps of 32. All loads are 32B vector loads of pre-packed frags.
// ---------------------------------------------------------------------------
__global__ void __launch_bounds__(256)
gemm_wmma_kernel(const _Float16* __restrict__ xf,
                 const _Float16* __restrict__ wf_r,
                 float* __restrict__ xw) {
    int wave = threadIdx.x >> 5;
    int lane = threadIdx.x & 31;
    int tile = blockIdx.x * 8 + wave;
    if (tile >= NT) return;                 // wave-uniform: EXEC stays all-1s

    v8f acc0 = {}, acc1 = {}, acc2 = {}, acc3 = {};
#pragma unroll
    for (int kt = 0; kt < 4; ++kt) {
        v16h a  = *(const v16h*)(xf + (((size_t)tile * 4 + kt) * 32 + lane) * 16);
        v16h b0 = *(const v16h*)(wf_r + (((kt * 4 + 0) * 32) + lane) * 16);
        v16h b1 = *(const v16h*)(wf_r + (((kt * 4 + 1) * 32) + lane) * 16);
        v16h b2 = *(const v16h*)(wf_r + (((kt * 4 + 2) * 32) + lane) * 16);
        v16h b3 = *(const v16h*)(wf_r + (((kt * 4 + 3) * 32) + lane) * 16);
        acc0 = __builtin_amdgcn_wmma_f32_16x16x32_f16(false, a, false, b0, (short)0, acc0, false, false);
        acc1 = __builtin_amdgcn_wmma_f32_16x16x32_f16(false, a, false, b1, (short)0, acc1, false, false);
        acc2 = __builtin_amdgcn_wmma_f32_16x16x32_f16(false, a, false, b2, (short)0, acc2, false, false);
        acc3 = __builtin_amdgcn_wmma_f32_16x16x32_f16(false, a, false, b3, (short)0, acc3, false, false);
    }

    // C/D layout: lane l, VGPR v -> M = v + 8*(l>>4), Nc = l&15 (per 16-col tile).
    int row0 = tile * 16 + ((lane >> 4) * 8);
    int col  = lane & 15;
#pragma unroll
    for (int v = 0; v < 8; ++v) {
        int row = row0 + v;
        if (row < N) {
            float* o = xw + (size_t)row * D_OUT;
            o[ 0 + col] = acc0[v];
            o[16 + col] = acc1[v];
            o[32 + col] = acc2[v];
            o[48 + col] = acc3[v];
        }
    }
}

// Zero-fill agg (graph-capture-safe, float4 stores).
__global__ void zero_kernel(float4* __restrict__ p, int n4) {
    int idx = blockIdx.x * blockDim.x + threadIdx.x;
    if (idx < n4) p[idx] = make_float4(0.f, 0.f, 0.f, 0.f);
}

// ---------------------------------------------------------------------------
// SpMM: agg[rows[e]] += vals[e] * xw[cols[e]]. One wave per edge, float2/lane
// (coalesced 256B gather), f32 global atomics for the scatter (L2-resident).
// ---------------------------------------------------------------------------
__global__ void __launch_bounds__(256)
spmm_kernel(const float* __restrict__ xw,
            const int*   __restrict__ rows,
            const int*   __restrict__ cols,
            const float* __restrict__ vals,
            float* __restrict__ agg) {
    int e    = blockIdx.x * 8 + (threadIdx.x >> 5);
    int lane = threadIdx.x & 31;
    if (e >= E) return;
    int   c = cols[e];
    int   rw = rows[e];
    float v = vals[e];
    float2 m = ((const float2*)(xw + (size_t)c * D_OUT))[lane];
    float* dst = agg + (size_t)rw * D_OUT + lane * 2;
    atomicAdd(dst,     m.x * v);
    atomicAdd(dst + 1, m.y * v);
}

// Epilogue per relation: total (in d_out) = [init ? 0 : total] + relu(agg + b_r).
__global__ void epilogue_kernel(const float* __restrict__ agg,
                                const float* __restrict__ b_r,
                                float* __restrict__ out, int init) {
    int idx = blockIdx.x * blockDim.x + threadIdx.x;
    if (idx >= N * D_OUT) return;
    float v = fmaxf(agg[idx] + b_r[idx & (D_OUT - 1)], 0.0f);
    if (init) out[idx] = v;
    else      out[idx] += v;
}

// Row-wise L2 normalize: one wave per row, float2/lane, wave shuffle reduce.
__global__ void normalize_kernel(float* __restrict__ out) {
    int row  = blockIdx.x * 8 + (threadIdx.x >> 5);
    int lane = threadIdx.x & 31;
    if (row >= N) return;
    float2* p = (float2*)(out + (size_t)row * D_OUT);
    float2 v = p[lane];
    float ss = v.x * v.x + v.y * v.y;
#pragma unroll
    for (int off = 16; off > 0; off >>= 1)
        ss += __shfl_xor(ss, off, 32);
    float inv = 1.0f / fmaxf(sqrtf(ss), 1e-12f);
    v.x *= inv; v.y *= inv;
    p[lane] = v;
}

extern "C" void kernel_launch(void* const* d_in, const int* in_sizes, int n_in,
                              void* d_out, int out_size, void* d_ws, size_t ws_size,
                              hipStream_t stream) {
    const float* x    = (const float*)d_in[0];
    const float* W    = (const float*)d_in[1];
    const float* b    = (const float*)d_in[2];
    const float* vals = (const float*)d_in[3];
    const int*   rows = (const int*)d_in[4];
    const int*   cols = (const int*)d_in[5];
    float* out = (float*)d_out;

    // Workspace carve-up (all offsets multiples of 256B). Total ~38.5 MB.
    char* ws = (char*)d_ws;
    _Float16* xf  = (_Float16*)ws;                                 // 12,800,000 B
    _Float16* wf  = (_Float16*)(ws + 12800000);                    //     65,536 B
    float*    xw  = (float*)   (ws + 12800000 + 65536);            // 12,800,000 B
    float*    agg = (float*)   (ws + 12800000 + 65536 + 12800000); // 12,800,000 B

    pack_x_kernel<<<(NT * 2048 + 255) / 256, 256, 0, stream>>>(x, xf);
    pack_w_kernel<<<(R * 8192) / 256, 256, 0, stream>>>(W, wf);

    for (int r = 0; r < R; ++r) {
        zero_kernel<<<(N * D_OUT / 4 + 255) / 256, 256, 0, stream>>>((float4*)agg, N * D_OUT / 4);
        gemm_wmma_kernel<<<(NT + 7) / 8, 256, 0, stream>>>(xf, wf + (size_t)r * 8192, xw);
        spmm_kernel<<<E / 8, 256, 0, stream>>>(xw,
                                               rows + (size_t)r * E,
                                               cols + (size_t)r * E,
                                               vals + (size_t)r * E,
                                               agg);
        epilogue_kernel<<<(N * D_OUT + 255) / 256, 256, 0, stream>>>(agg, b + r * D_OUT, out, r == 0);
    }
    normalize_kernel<<<(N + 7) / 8, 256, 0, stream>>>(out);
}
